// HATR_I_65317862637883
// MI455X (gfx1250) — matrix operand order repaired and checked by reading, not compiled, not run on hardware
//
#include <hip/hip_runtime.h>

#define N_NODES 4000
#define S_LEN   32
#define D_DIM   128
#define SD      4096          // S*D
#define L_LAYERS 2
#define C_CLS   10
#define KDIM    4000          // GEMM contraction length (nodes); also XT row stride

// ---------------------------------------------------------------------------
// CDNA5 WMMA types
// ---------------------------------------------------------------------------
typedef __attribute__((ext_vector_type(16))) __bf16 bf16x16;
typedef __attribute__((ext_vector_type(8)))  float  f32x8;

struct U128 { unsigned int x, y, z, w; };
struct U64  { unsigned int x, y; };
union Frag { U128 q[2]; bf16x16 m; };

__device__ __forceinline__ unsigned short f2bf(float x) {
    union { float f; unsigned int u; } v; v.f = x;
    unsigned int r = v.u + 0x7FFFu + ((v.u >> 16) & 1u);   // round-nearest-even
    return (unsigned short)(r >> 16);
}

// ---------------------------------------------------------------------------
// Big bf16 WMMA GEMM:  C[4000,4096] (f32) = P[4000,4000] (bf16) * X (bf16)
// X is supplied PRE-TRANSPOSED: XT[4096, 4000] so B staging is a straight copy.
// Block: 256 threads = 8 waves; macro-tile 160(M) x 128(N), K-step 32.
// Waves: 2 M-groups (5 tiles) x 4 N-groups (2 tiles) -> 10 WMMA / 7 frag loads.
// LDS double-buffered; next K-tile global loads issued before the WMMA chain.
// ---------------------------------------------------------------------------
__global__ __launch_bounds__(256) void k_gemm_bf16(
    const unsigned short* __restrict__ Pm,   // [4000,4000] bf16 row-major
    const unsigned short* __restrict__ XT,   // [4096,4000] bf16 row-major (X^T)
    float* __restrict__ Cm)                  // [4000,4096] f32 row-major
{
    __shared__ __align__(16) unsigned short As[2][160 * 40];  // [m][k], stride 80B
    __shared__ __align__(16) unsigned short Bs[2][128 * 40];  // [n][k], stride 80B
    const int tid  = threadIdx.x;
    const int m0   = blockIdx.y * 160;
    const int n0   = blockIdx.x * 128;
    const int wv   = tid >> 5;
    const int lane = tid & 31;
    const int lm   = lane & 15;
    const int hh   = lane >> 4;
    const int mg   = wv & 1;          // m-tiles mg*5 .. mg*5+4
    const int ng   = wv >> 1;         // n-tiles ng*2, ng*2+1

    const int srow = tid >> 3;        // staging row 0..31 (+i*32)
    const int sseg = tid & 7;         // staging k-segment (sseg*4 bf16 = 8B)

    f32x8 acc[5][2];
    for (int i = 0; i < 5; ++i)
        for (int j2 = 0; j2 < 2; ++j2)
            for (int r = 0; r < 8; ++r) acc[i][j2][r] = 0.f;

    U64 aR[5], bR[4];
    auto stageLoad = [&](int kk) {
        const int kb = kk * 32 + sseg * 4;
#pragma unroll
        for (int i = 0; i < 5; ++i)
            aR[i] = *(const U64*)&Pm[(size_t)(m0 + srow + i * 32) * KDIM + kb];
#pragma unroll
        for (int i = 0; i < 4; ++i)
            bR[i] = *(const U64*)&XT[(size_t)(n0 + srow + i * 32) * KDIM + kb];
    };
    auto stageStore = [&](int buf) {
#pragma unroll
        for (int i = 0; i < 5; ++i)
            *(U64*)&As[buf][(srow + i * 32) * 40 + sseg * 4] = aR[i];
#pragma unroll
        for (int i = 0; i < 4; ++i)
            *(U64*)&Bs[buf][(srow + i * 32) * 40 + sseg * 4] = bR[i];
    };

    stageLoad(0);
    stageStore(0);
    __syncthreads();

    for (int kk = 0; kk < KDIM / 32; ++kk) {
        const int cur = kk & 1;
        if (kk + 1 < KDIM / 32) stageLoad(kk + 1);   // global loads in flight

        // ---- gather ALL fragments first (batched ds loads, single wait) ----
        Frag bfr[2], afr[5];
#pragma unroll
        for (int j2 = 0; j2 < 2; ++j2) {
            // B frag: lane = N (lm); elements e -> K = e + 16*hh (contiguous 32B)
            const unsigned short* p = &Bs[cur][(ng * 32 + j2 * 16 + lm) * 40 + hh * 16];
            bfr[j2].q[0] = *(const U128*)(p);
            bfr[j2].q[1] = *(const U128*)(p + 8);
        }
#pragma unroll
        for (int i = 0; i < 5; ++i) {
            // A frag: lane = M (lm); elements e -> K = e + 16*hh (contiguous 32B)
            const unsigned short* p = &As[cur][(mg * 80 + i * 16 + lm) * 40];
            afr[i].q[0] = *(const U128*)(p + 16 * hh);
            afr[i].q[1] = *(const U128*)(p + 16 * hh + 8);
        }
        // ---- 10 back-to-back WMMAs ----
#pragma unroll
        for (int i = 0; i < 5; ++i)
#pragma unroll
            for (int j2 = 0; j2 < 2; ++j2)
                acc[i][j2] = __builtin_amdgcn_wmma_f32_16x16x32_bf16(
                    false, afr[i].m, false, bfr[j2].m, (short)0, acc[i][j2], false, false);

        if (kk + 1 < KDIM / 32) stageStore(cur ^ 1); // fill other buffer
        __syncthreads();
    }

    // D layout: VGPR r, lanes0-15 -> M=r, lanes16-31 -> M=8+r ; N = lane%16
    for (int i = 0; i < 5; ++i)
        for (int j2 = 0; j2 < 2; ++j2)
            for (int r = 0; r < 8; ++r)
                Cm[(size_t)(m0 + mg * 80 + i * 16 + hh * 8 + r) * SD
                   + n0 + ng * 32 + j2 * 16 + lm] = acc[i][j2][r];
}

// ---------------------------------------------------------------------------
// Graph normalization: s = (rowsum(A)+1)^-1/2, then P_f / P_v scale factors
// ---------------------------------------------------------------------------
__global__ void k_zero(float* p, int n) {
    int i = blockIdx.x * 256 + threadIdx.x;
    if (i < n) p[i] = 0.f;
}

__global__ __launch_bounds__(256) void k_deg(const float* __restrict__ A, float* __restrict__ s) {
    __shared__ float red[256];
    int m = blockIdx.x;
    float acc = 0.f;
    for (int n = threadIdx.x; n < N_NODES; n += 256) acc += A[(size_t)m * N_NODES + n];
    red[threadIdx.x] = acc; __syncthreads();
    for (int st = 128; st > 0; st >>= 1) {
        if (threadIdx.x < st) red[threadIdx.x] += red[threadIdx.x + st];
        __syncthreads();
    }
    if (threadIdx.x == 0) s[m] = rsqrtf(red[0] + 1.0f);
}

__global__ __launch_bounds__(256) void k_rowW(const float* __restrict__ A,
                                              const float* __restrict__ s,
                                              float* __restrict__ rowW) {
    __shared__ float red[256];
    int m = blockIdx.x;
    float acc = 0.f;
    for (int n = threadIdx.x; n < N_NODES; n += 256)
        acc += A[(size_t)m * N_NODES + n] * s[n];
    red[threadIdx.x] = acc; __syncthreads();
    for (int st = 128; st > 0; st >>= 1) {
        if (threadIdx.x < st) red[threadIdx.x] += red[threadIdx.x + st];
        __syncthreads();
    }
    if (threadIdx.x == 0) rowW[m] = red[0];
}

__global__ __launch_bounds__(128) void k_colW(const float* __restrict__ A,
                                              const float* __restrict__ s,
                                              float* __restrict__ colW) {
    int col = blockIdx.x * 128 + threadIdx.x;
    if (col >= N_NODES) return;
    int mBeg = blockIdx.y * 500, mEnd = mBeg + 500;
    float acc = 0.f;
    for (int m = mBeg; m < mEnd; ++m) acc += A[(size_t)m * N_NODES + col] * s[m];
    atomicAdd(&colW[col], acc);
}

__global__ void k_finish(const float* s, const float* rowW, const float* colW,
                         float* rinv, float* cinv) {
    int i = blockIdx.x * 256 + threadIdx.x;
    if (i < N_NODES) {
        rinv[i] = 1.f / (s[i] * (rowW[i] + s[i]));
        cinv[i] = 1.f / (s[i] * (colW[i] + s[i]));
    }
}

// Pf[m,n] = (A[m,n]+I) * s_m * s_n * rinv[m]   (bf16, coalesced)
// Pv[n,m] = (A[m,n]+I) * s_m * s_n * cinv[m]   (bf16, via LDS tile transpose)
__global__ __launch_bounds__(256) void k_makeP(const float* __restrict__ A,
                                               const float* __restrict__ s,
                                               const float* __restrict__ rinv,
                                               const float* __restrict__ cinv,
                                               unsigned short* __restrict__ Pf,
                                               unsigned short* __restrict__ Pv) {
    __shared__ float pvt[64][65];
    int m0 = blockIdx.y * 64, n0 = blockIdx.x * 64;
    int j = threadIdx.x & 63, i0 = threadIdx.x >> 6;
    for (int st = 0; st < 16; ++st) {
        int i = i0 + st * 4;
        int m = m0 + i, n = n0 + j;
        if (m < N_NODES && n < N_NODES) {
            float a = A[(size_t)m * N_NODES + n] + ((m == n) ? 1.f : 0.f);
            float base = a * s[m] * s[n];
            Pf[(size_t)m * N_NODES + n] = f2bf(base * rinv[m]);
            pvt[j][i] = base * cinv[m];
        }
    }
    __syncthreads();
    for (int st = 0; st < 16; ++st) {
        int i = i0 + st * 4;
        int nn = n0 + i, mm = m0 + j;
        if (nn < N_NODES && mm < N_NODES)
            Pv[(size_t)nn * N_NODES + mm] = f2bf(pvt[i][j]);
    }
}

// ---------------------------------------------------------------------------
// Producers of the transposed bf16 GEMM input XT[4096, 4000]
// ---------------------------------------------------------------------------
// XT[s*128+d][n] = embs[l,s,n,d]  (LDS 64x64 tile transpose, both sides coalesced)
__global__ __launch_bounds__(256) void k_castX_T(const float* __restrict__ embs,
                                                 unsigned short* __restrict__ XT, int l) {
    __shared__ float tr[64][65];
    int n0 = blockIdx.x * 64;
    int c0 = blockIdx.y * 64;
    int sIdx = c0 >> 7;
    int dbase = c0 & 127;
    int j = threadIdx.x & 63, i0 = threadIdx.x >> 6;
    const float* src = embs + ((size_t)l * S_LEN + sIdx) * N_NODES * D_DIM;
    for (int st = 0; st < 16; ++st) {
        int i = i0 + st * 4;                       // n-local
        if (n0 + i < N_NODES)
            tr[j][i] = src[(size_t)(n0 + i) * D_DIM + dbase + j];  // tr[d][n]
    }
    __syncthreads();
    for (int st = 0; st < 16; ++st) {
        int i = i0 + st * 4;                       // c-local
        if (n0 + j < N_NODES)
            XT[(size_t)(c0 + i) * KDIM + n0 + j] = f2bf(tr[i][j]);
    }
}

// XT[c][n] = V[n][c]  (f32 [4000,4096] -> bf16 transposed)
__global__ __launch_bounds__(256) void k_castBF_T(const float* __restrict__ V,
                                                  unsigned short* __restrict__ XT) {
    __shared__ float tr[64][65];
    int n0 = blockIdx.x * 64;
    int c0 = blockIdx.y * 64;
    int j = threadIdx.x & 63, i0 = threadIdx.x >> 6;
    for (int st = 0; st < 16; ++st) {
        int i = i0 + st * 4;                       // n-local
        if (n0 + i < N_NODES)
            tr[j][i] = V[(size_t)(n0 + i) * SD + c0 + j];          // tr[c][n]
    }
    __syncthreads();
    for (int st = 0; st < 16; ++st) {
        int i = i0 + st * 4;                       // c-local
        if (n0 + j < N_NODES)
            XT[(size_t)(c0 + i) * KDIM + n0 + j] = f2bf(tr[i][j]);
    }
}

// ---------------------------------------------------------------------------
// Per-(n,s)-row projections
// ---------------------------------------------------------------------------
// h1 = relu(U@W0 + V@W1) -> bf16, written TRANSPOSED (feeds next GEMM round)
__global__ __launch_bounds__(128) void k_proj_relu(
    const float* __restrict__ U, const float* __restrict__ V,
    const float* __restrict__ Wk, unsigned short* __restrict__ outT) {
    __shared__ float u[128], v[128];
    int row = blockIdx.x, t = threadIdx.x;
    int n = row >> 5, sIdx = row & 31;
    size_t base = (size_t)row * D_DIM;
    u[t] = U[base + t]; v[t] = V[base + t]; __syncthreads();
    const float* W0 = Wk;
    const float* W1 = Wk + D_DIM * D_DIM;
    float acc = 0.f;
    for (int d = 0; d < D_DIM; ++d)
        acc += u[d] * W0[d * D_DIM + t] + v[d] * W1[d * D_DIM + t];
    outT[(size_t)(sIdx * D_DIM + t) * KDIM + n] = f2bf(fmaxf(acc, 0.f));
}

// h2 = U@W0 + V@W1 ; fused = [0.5*h2, 0.5*X] @ W_psi + b_psi  (in-place into U)
__global__ __launch_bounds__(128) void k_proj_fuse(
    const float* Uin, const float* __restrict__ V,
    const float* __restrict__ embs, int l,
    const float* __restrict__ Wk, const float* __restrict__ Wpsi,
    const float* __restrict__ bpsi, float* fusedOut) {
    __shared__ float u[128], v[128], h2[128], xx[128];
    int row = blockIdx.x, t = threadIdx.x;
    int n = row >> 5, sIdx = row & 31;
    size_t base = (size_t)row * D_DIM;
    u[t] = Uin[base + t]; v[t] = V[base + t];
    xx[t] = embs[(((size_t)l * S_LEN + sIdx) * N_NODES + n) * D_DIM + t];
    __syncthreads();
    const float* W0 = Wk;
    const float* W1 = Wk + D_DIM * D_DIM;
    float acc = 0.f;
    for (int d = 0; d < D_DIM; ++d)
        acc += u[d] * W0[d * D_DIM + t] + v[d] * W1[d * D_DIM + t];
    h2[t] = acc; __syncthreads();
    float f = bpsi[t];
    for (int d = 0; d < D_DIM; ++d)
        f += 0.5f * h2[d] * Wpsi[d * D_DIM + t]
           + 0.5f * xx[d] * Wpsi[(D_DIM + d) * D_DIM + t];
    fusedOut[base + t] = f;
}

// ---------------------------------------------------------------------------
// Attention head (one block per node)
// ---------------------------------------------------------------------------
__global__ __launch_bounds__(128) void k_attn(
    const float* __restrict__ fused, const float* __restrict__ linIn,
    const float* __restrict__ linOut, const float* __restrict__ ae,
    const float* __restrict__ ab, float* __restrict__ outp) {
    __shared__ float fl[128], q[128], sc[32], comb[256];
    __shared__ float mxv, smv;
    int n = blockIdx.x, t = threadIdx.x;
    size_t rb = (size_t)n * S_LEN * D_DIM;
    fl[t] = fused[rb + 31 * D_DIM + t];
    __syncthreads();
    float qa = 0.f;
    for (int j = 0; j < 128; ++j) qa += fl[j] * linIn[t * 128 + j];
    q[t] = qa; __syncthreads();
    if (t < 32) {
        float sv = 0.f;
        const float* frow = &fused[rb + (size_t)t * D_DIM];
        for (int j = 0; j < 128; ++j) sv += q[j] * frow[j];
        sc[t] = sv;
    }
    __syncthreads();
    if (t == 0) { float mx = sc[0]; for (int i = 1; i < 32; ++i) mx = fmaxf(mx, sc[i]); mxv = mx; }
    __syncthreads();
    if (t < 32) sc[t] = expf(sc[t] - mxv);
    __syncthreads();
    if (t == 0) { float sm = 0.f; for (int i = 0; i < 32; ++i) sm += sc[i]; smv = 1.f / sm; }
    __syncthreads();
    float aen = ae[n], abn = ab[n];
    float acc = 0.f;
    for (int sIdx = 0; sIdx < 32; ++sIdx) {
        float attw = sc[sIdx] * smv;
        float mix  = attw * fused[rb + (size_t)sIdx * D_DIM + t];
        float bt   = expf(-abn * (float)(31 - sIdx));
        float t2   = fmaxf(aen * mix * bt, 0.f);
        acc += mix + t2;
    }
    comb[t] = acc; comb[128 + t] = q[t]; __syncthreads();
    float o = 0.f;
    for (int j = 0; j < 256; ++j) o += comb[j] * linOut[t * 256 + j];
    outp[(size_t)n * D_DIM + t] = tanhf(o);
}

// ---------------------------------------------------------------------------
// Betas / combination / classifier head
// ---------------------------------------------------------------------------
__global__ void k_tanhK(const float* __restrict__ src, float* __restrict__ dst, int n) {
    int gid = blockIdx.x * 256 + threadIdx.x;
    if (gid < n) dst[gid] = tanhf(src[gid]);
}

__global__ __launch_bounds__(128) void k_G(const float* __restrict__ emb,
                                           const float* __restrict__ touts,
                                           float* __restrict__ G) {
    int b = blockIdx.x; int l = b >> 7; int d = b & 127; int e = threadIdx.x;
    float acc = 0.f;
    for (int nn = 0; nn < N_NODES; ++nn) {
        float qv = fmaxf(emb[(size_t)nn * 128 + d], 0.f);
        acc += qv * touts[((size_t)l * N_NODES + nn) * 128 + e];
    }
    G[((size_t)l * 128 + d) * 128 + e] = acc;
}

__global__ void k_bnorm(const float* __restrict__ G, float* __restrict__ Bn) {
    int gid = blockIdx.x * 256 + threadIdx.x;
    if (gid >= 128 * 128) return;
    float g0 = G[gid], g1 = G[16384 + gid];
    float mx = fmaxf(g0, g1);
    float e0 = expf(g0 - mx), e1 = expf(g1 - mx);
    float inv = 1.f / (e0 + e1);
    Bn[gid] = e0 * inv; Bn[16384 + gid] = e1 * inv;
}

__global__ __launch_bounds__(128) void k_Zf(const float* __restrict__ outp,
                                            const float* __restrict__ Bn,
                                            float* __restrict__ Z) {
    __shared__ float o0[128], o1[128];
    int n = blockIdx.x, e = threadIdx.x;
    o0[e] = outp[(size_t)n * 128 + e];
    o1[e] = outp[((size_t)N_NODES + n) * 128 + e];
    __syncthreads();
    float acc = 0.f;
    for (int d = 0; d < 128; ++d)
        acc += o0[d] * Bn[d * 128 + e] + o1[d] * Bn[16384 + d * 128 + e];
    Z[(size_t)n * 128 + e] = acc;
}

__global__ __launch_bounds__(128) void k_Pmat(const float* __restrict__ Z,
                                              const float* __restrict__ Wp,
                                              float* __restrict__ Pmat) {
    __shared__ float z[128], lg[10];
    int n = blockIdx.x, t = threadIdx.x;
    z[t] = Z[(size_t)n * 128 + t]; __syncthreads();
    if (t < 10) {
        float a = 0.f;
        for (int d = 0; d < 128; ++d) a += z[d] * Wp[d * 10 + t];
        lg[t] = a;
    }
    __syncthreads();
    if (t == 0) {
        float mx = lg[0];
        for (int c = 1; c < 10; ++c) mx = fmaxf(mx, lg[c]);
        float e[10]; float sm = 0.f;
        for (int c = 0; c < 10; ++c) { e[c] = expf(lg[c] - mx); sm += e[c]; }
        float inv = 1.f / sm;
        for (int c = 0; c < 10; ++c) Pmat[n * 10 + c] = e[c] * inv;
    }
}

__global__ __launch_bounds__(256) void k_AP(const float* __restrict__ A,
                                            const float* __restrict__ Pmat,
                                            float* __restrict__ AP) {
    __shared__ float part[256 * 10];
    int n = blockIdx.x, t = threadIdx.x;
    float acc[10];
    for (int c = 0; c < 10; ++c) acc[c] = 0.f;
    for (int m = t; m < N_NODES; m += 256) {
        float a = A[(size_t)n * N_NODES + m];
        for (int c = 0; c < 10; ++c) acc[c] += a * Pmat[m * 10 + c];
    }
    for (int c = 0; c < 10; ++c) part[t * 10 + c] = acc[c];
    __syncthreads();
    if (t < 10) {
        float s = 0.f;
        for (int i = 0; i < 256; ++i) s += part[i * 10 + t];
        AP[n * 10 + t] = s;
    }
}

__global__ void k_small(const float* __restrict__ Pmat, const float* __restrict__ AP,
                        const float* __restrict__ Z, float* __restrict__ eta,
                        float* __restrict__ Cmat) {
    int id = blockIdx.x * 256 + threadIdx.x;
    if (id < 100) {
        int c1 = id / 10, c2 = id % 10; float a = 0.f;
        for (int nn = 0; nn < N_NODES; ++nn) a += Pmat[nn * 10 + c1] * AP[nn * 10 + c2];
        eta[id] = a;
    } else if (id < 100 + C_CLS * D_DIM) {
        int j = id - 100; int c = j >> 7, d = j & 127; float a = 0.f;
        for (int nn = 0; nn < N_NODES; ++nn) a += Pmat[nn * 10 + c] * Z[(size_t)nn * 128 + d];
        Cmat[j] = a;
    }
}

__global__ __launch_bounds__(128) void k_cprime(const float* __restrict__ Cmat,
                                                const float* __restrict__ Wc,
                                                const float* __restrict__ eta,
                                                float* __restrict__ Cp) {
    __shared__ float tmp[10 * 128];
    int i = threadIdx.x;
    for (int c = 0; c < 10; ++c) {
        float a = 0.f;
        for (int d = 0; d < 128; ++d) a += Cmat[c * 128 + d] * Wc[i * 128 + d];
        tmp[c * 128 + i] = a;
    }
    __syncthreads();
    for (int c = 0; c < 10; ++c) {
        float a = 0.f;
        for (int c2 = 0; c2 < 10; ++c2) a += eta[c * 10 + c2] * tmp[c2 * 128 + i];
        Cp[c * 128 + i] = fmaxf(a, 0.f);
    }
}

__global__ void k_out(const float* __restrict__ Pmat, const float* __restrict__ Cp,
                      float* __restrict__ out) {
    int gid = blockIdx.x * 256 + threadIdx.x;
    if (gid >= N_NODES * D_DIM) return;
    int n = gid >> 7, i = gid & 127;
    float a = 0.f;
    for (int c = 0; c < 10; ++c) a += Pmat[n * 10 + c] * Cp[c * 128 + i];
    out[gid] = a;
}

// ---------------------------------------------------------------------------
// Orchestration
// ---------------------------------------------------------------------------
extern "C" void kernel_launch(void* const* d_in, const int* in_sizes, int n_in,
                              void* d_out, int out_size, void* d_ws, size_t ws_size,
                              hipStream_t stream) {
    (void)in_sizes; (void)n_in; (void)out_size; (void)ws_size;
    const float* A      = (const float*)d_in[0];
    const float* embs   = (const float*)d_in[1];
    const float* conv1  = (const float*)d_in[2];
    const float* conv2  = (const float*)d_in[3];
    const float* linIn  = (const float*)d_in[4];
    const float* linOut = (const float*)d_in[5];
    const float* Wpsi   = (const float*)d_in[6];
    const float* bpsi   = (const float*)d_in[7];
    const float* ae     = (const float*)d_in[8];
    const float* ab     = (const float*)d_in[9];
    const float* emb    = (const float*)d_in[10];
    const float* Wp     = (const float*)d_in[11];
    const float* Wc     = (const float*)d_in[12];
    float* out = (float*)d_out;

    char* wsp = (char*)d_ws; size_t off = 0;
    auto alloc = [&](size_t bytes) -> void* {
        void* p = wsp + off;
        off = (off + bytes + 255) & ~(size_t)255;
        return p;
    };
    float* s    = (float*)alloc((size_t)N_NODES * 4);
    float* rowW = (float*)alloc((size_t)N_NODES * 4);
    float* colW = (float*)alloc((size_t)N_NODES * 4);
    float* rinv = (float*)alloc((size_t)N_NODES * 4);
    float* cinv = (float*)alloc((size_t)N_NODES * 4);
    unsigned short* Pf = (unsigned short*)alloc((size_t)N_NODES * N_NODES * 2);
    unsigned short* Pv = (unsigned short*)alloc((size_t)N_NODES * N_NODES * 2);
    unsigned short* XT = (unsigned short*)alloc((size_t)SD * KDIM * 2);
    float* U       = (float*)alloc((size_t)N_NODES * SD * 4);
    float* V       = (float*)alloc((size_t)N_NODES * SD * 4);
    float* outputs = (float*)alloc((size_t)L_LAYERS * N_NODES * D_DIM * 4);
    float* touts   = (float*)alloc((size_t)L_LAYERS * N_NODES * D_DIM * 4);
    float* G    = (float*)alloc(2 * 128 * 128 * 4);
    float* Bn   = (float*)alloc(2 * 128 * 128 * 4);
    float* Zf   = (float*)alloc((size_t)N_NODES * 128 * 4);
    float* Pm   = (float*)alloc((size_t)N_NODES * 10 * 4);
    float* AP   = (float*)alloc((size_t)N_NODES * 10 * 4);
    float* eta  = (float*)alloc(100 * 4);
    float* Cmat = (float*)alloc(10 * 128 * 4);
    float* Cp   = (float*)alloc(10 * 128 * 4);

    // ---- graph normalization / P_f, P_v (bf16) ----
    k_zero  <<<16, 256, 0, stream>>>(colW, N_NODES);
    k_deg   <<<N_NODES, 256, 0, stream>>>(A, s);
    k_rowW  <<<N_NODES, 256, 0, stream>>>(A, s, rowW);
    k_colW  <<<dim3(32, 8), 128, 0, stream>>>(A, s, colW);
    k_finish<<<16, 256, 0, stream>>>(s, rowW, colW, rinv, cinv);
    k_makeP <<<dim3(63, 63), 256, 0, stream>>>(A, s, rinv, cinv, Pf, Pv);

    const dim3 gGrid(32, 25);     // 4096/128 x 4000/160
    const dim3 tGrid(63, 64);     // n-tiles x c-tiles for 64x64 transposing casts
    for (int l = 0; l < L_LAYERS; ++l) {
        // conv1 diffusion: U = Pf@X ; V = Pv@(Pv@X)
        k_castX_T<<<tGrid, 256, 0, stream>>>(embs, XT, l);
        k_gemm_bf16<<<gGrid, 256, 0, stream>>>(Pf, XT, U);
        k_gemm_bf16<<<gGrid, 256, 0, stream>>>(Pv, XT, V);
        k_castBF_T<<<tGrid, 256, 0, stream>>>(V, XT);
        k_gemm_bf16<<<gGrid, 256, 0, stream>>>(Pv, XT, V);
        k_proj_relu<<<N_NODES * S_LEN, 128, 0, stream>>>(U, V, conv1, XT); // h1 -> XT
        // conv2 diffusion on h1
        k_gemm_bf16<<<gGrid, 256, 0, stream>>>(Pf, XT, U);
        k_gemm_bf16<<<gGrid, 256, 0, stream>>>(Pv, XT, V);
        k_castBF_T<<<tGrid, 256, 0, stream>>>(V, XT);
        k_gemm_bf16<<<gGrid, 256, 0, stream>>>(Pv, XT, V);
        // h2 + theta-fusion with raw X -> fused (in-place into U)
        k_proj_fuse<<<N_NODES * S_LEN, 128, 0, stream>>>(U, V, embs, l, conv2, Wpsi, bpsi, U);
        k_attn<<<N_NODES, 128, 0, stream>>>(U, linIn, linOut, ae, ab,
                                            outputs + (size_t)l * N_NODES * D_DIM);
    }

    // ---- betas / Z / classifier head ----
    k_tanhK <<<4000, 256, 0, stream>>>(outputs, touts, L_LAYERS * N_NODES * D_DIM);
    k_G     <<<256, 128, 0, stream>>>(emb, touts, G);
    k_bnorm <<<64, 256, 0, stream>>>(G, Bn);
    k_Zf    <<<N_NODES, 128, 0, stream>>>(outputs, Bn, Zf);
    k_Pmat  <<<N_NODES, 128, 0, stream>>>(Zf, Wp, Pm);
    k_AP    <<<N_NODES, 256, 0, stream>>>(A, Pm, AP);
    k_small <<<6, 256, 0, stream>>>(Pm, AP, Zf, eta, Cmat);
    k_cprime<<<1, 128, 0, stream>>>(Cmat, Wc, eta, Cp);
    k_out   <<<2000, 256, 0, stream>>>(Pm, Cp, out);
}
